// OPTMLSTMCellTorch_29832842838106
// MI455X (gfx1250) — compile-verified, matrix operand order checked by reading, not run on hardware
//
#include <hip/hip_runtime.h>

typedef __attribute__((ext_vector_type(16))) _Float16 v16h;
typedef __attribute__((ext_vector_type(8)))  _Float16 v8h;
typedef __attribute__((ext_vector_type(8)))  float    v8f;
typedef int v4i __attribute__((vector_size(16)));
typedef __attribute__((address_space(1))) v4i* gv4p;   // global ptr to 16B vector
typedef __attribute__((address_space(3))) v4i* lv4p;   // LDS ptr to 16B vector

#define B_ROWS   65536
#define HDIM     128
#define GV_COLS  768
#define GD_ITERS 13
#define S_CHUNKS 16
#define NBLK     21
#define K2_LDK   40   // LDS pitch in halfs: 80B rows -> 16B-aligned segs, conflict-free banks

// symmetric 6x6 block enumeration (bi <= bj)
__constant__ int c_bi[NBLK] = {0,0,0,0,0,0, 1,1,1,1,1, 2,2,2,2, 3,3,3, 4,4, 5};
__constant__ int c_bj[NBLK] = {0,1,2,3,4,5, 1,2,3,4,5, 2,3,4,5, 3,4,5, 4,5, 5};

// ---- CDNA5 async global->LDS path (ASYNCcnt), guarded for toolchain drift ----
#if defined(__has_builtin)
#if __has_builtin(__builtin_amdgcn_global_load_async_to_lds_b128)
#define USE_ASYNC_LDS 1
#endif
#if __has_builtin(__builtin_amdgcn_s_wait_asynccnt)
#define USE_WAIT_ASYNC_BUILTIN 1
#endif
#endif

__device__ __forceinline__ void async_copy16(const _Float16* g, _Float16* l) {
#if defined(USE_ASYNC_LDS)
  __builtin_amdgcn_global_load_async_to_lds_b128(
      (gv4p)(unsigned long long)g,
      (lv4p)(unsigned long long)l,
      0, 0);
#else
  *(v8h*)l = *(const v8h*)g;  // vectorized synchronous fallback
#endif
}

__device__ __forceinline__ void wait_async_all() {
#if defined(USE_ASYNC_LDS)
#if defined(USE_WAIT_ASYNC_BUILTIN)
  __builtin_amdgcn_s_wait_asynccnt(0);
#else
  asm volatile("s_wait_asynccnt 0x0" ::: "memory");
#endif
#endif
}

// Load one 16x32 f16 A/B fragment (v16h) from LDS where K is contiguous per row.
// Per CDNA5 ISA 7.12.2: lane L (L<16) holds row M=L, K in {0..7,16..23};
// lane L+16 holds row M=L, K in {8..15,24..31}. rowp = &lds[M * ldk], kh = lane>>4.
__device__ __forceinline__ v16h frag_k32(const _Float16* rowp, int kh) {
  union { v16h v; unsigned u[8]; } r;
  const unsigned* p32 = (const unsigned*)rowp;  // 2 halfs per dword, K pairs contiguous
#pragma unroll
  for (int q = 0; q < 8; ++q) {
    int kb = (q < 4) ? (kh * 8 + 2 * q) : (16 + kh * 8 + 2 * (q - 4));
    r.u[q] = p32[kb >> 1];
  }
  return r.v;
}

__device__ __forceinline__ float sigmoidf_(float v) { return 1.0f / (1.0f + __expf(-v)); }

// ---------------------------------------------------------------------------
// k0: pack combined weight Wc(512x256) = [W_ih | W_hh] into per-lane f16
// B-matrix fragments: wpack[((tile*8+ks)*32+lane)*16 + e], tile over N=512/16.
// ---------------------------------------------------------------------------
__global__ __launch_bounds__(256) void k0_pack(const float* __restrict__ W_ih,
                                               const float* __restrict__ W_hh,
                                               _Float16* __restrict__ wpack) {
  int id = blockIdx.x * 256 + threadIdx.x;  // 32*8*32*16 = 131072
  int e = id & 15, lane = (id >> 4) & 31, ks = (id >> 9) & 7, t = id >> 12;
  int n  = t * 16 + (lane & 15);
  int kh = lane >> 4;
  int p  = e >> 1;
  int kb = (p < 4) ? (kh * 8 + 2 * p) : (16 + kh * 8 + 2 * (p - 4));
  int k  = ks * 32 + kb + (e & 1);
  float v = (k < 128) ? W_ih[n * 128 + k] : W_hh[n * 128 + (k - 128)];
  wpack[id] = (_Float16)v;
}

// ---------------------------------------------------------------------------
// k1: fused LSTM cell. Per WG: 128 rows x 512 cols of z via WMMA (B-fragment
// loads software-pipelined), gates, c_new -> d_out (NT), gv(f32) -> ws,
// gvT(f16, transposed, for the async Gram pass) -> ws, r-partials via LDS.
// ---------------------------------------------------------------------------
__global__ __launch_bounds__(256)
void k1_cell(const float* __restrict__ x, const float* __restrict__ h,
             const float* __restrict__ c, const float* __restrict__ b_ih,
             const _Float16* __restrict__ wpack,
             float* __restrict__ out_cnew, float* __restrict__ gv,
             _Float16* __restrict__ gvT, float* __restrict__ rpart) {
  __shared__ _Float16 Ash[128 * 132];  // 33 KB
  __shared__ float rsh[GV_COLS];       // 3 KB
  const int tid = threadIdx.x;
  const int lane = tid & 31, w = tid >> 5;
  const int kh = lane >> 4, nn = lane & 15;
  const int rowbase = blockIdx.x * 128;

  for (int j = tid; j < GV_COLS; j += 256) rsh[j] = 0.0f;

  v8f acc[4][8] = {};  // 4 gates x 8 col-tiles, 16 rows per wave

  for (int kc = 0; kc < 2; ++kc) {  // K half 0: x_feat, half 1: h
    __syncthreads();
    const float* src = (kc == 0) ? x : h;
    const int stride = (kc == 0) ? 129 : 128;
#pragma unroll
    for (int i = 0; i < 64; ++i) {
      int id = tid + i * 256;
      int rr = id >> 7, cc = id & 127;
      Ash[rr * 132 + cc] = (_Float16)src[(size_t)(rowbase + rr) * stride + cc];
    }
    __syncthreads();
#pragma unroll
    for (int ks4 = 0; ks4 < 4; ++ks4) {
      int ks = kc * 4 + ks4;
      v16h a = frag_k32(&Ash[(w * 16 + nn) * 132 + ks4 * 32], kh);
      // double-buffer B fragments across the 32 tiles to keep a load in flight
      v16h bc = *(const v16h*)&wpack[(((size_t)0 * 8 + ks) * 32 + lane) * 16];
#pragma unroll
      for (int tile = 0; tile < 32; ++tile) {
        int nxt = (tile < 31) ? (tile + 1) : tile;
        v16h bn = *(const v16h*)&wpack[(((size_t)nxt * 8 + ks) * 32 + lane) * 16];
        acc[tile >> 3][tile & 7] = __builtin_amdgcn_wmma_f32_16x16x32_f16(
            false, a, false, bc, (short)0, acc[tile >> 3][tile & 7], false, false);
        bc = bn;
      }
    }
  }

  // epilogue: gates + candidates
  float guar[8];
#pragma unroll
  for (int r = 0; r < 8; ++r) {
    int row = rowbase + w * 16 + r + 8 * kh;  // C layout: M = vgpr + 8*(lane>=16)
    guar[r] = x[(size_t)row * 129 + 128];
  }
  const int row0 = rowbase + w * 16 + 8 * kh;  // 8 consecutive rows per lane
#pragma unroll
  for (int t = 0; t < 8; ++t) {
    int hcol = t * 16 + nn;
    float b0 = b_ih[hcol], b1 = b_ih[128 + hcol], b2 = b_ih[256 + hcol], b3 = b_ih[384 + hcol];
    v8h pig, pfg, pct, pcn, pog, pht;
#pragma unroll
    for (int r = 0; r < 8; ++r) {
      int row = row0 + r;
      float ig = sigmoidf_(acc[0][t][r] + b0);
      float fg = sigmoidf_(acc[1][t][r] + b1);
      float ct = tanhf(acc[2][t][r] + b2);
      float og = sigmoidf_(acc[3][t][r] + b3);
      float cv = c[(size_t)row * HDIM + hcol];
      float cn = fg * cv + ig * ct;
      float ht = og * tanhf(cn);
      __builtin_nontemporal_store(cn, &out_cnew[(size_t)row * HDIM + hcol]);
      size_t gb = (size_t)row * GV_COLS + hcol;
      gv[gb]       = ig; gv[gb + 128] = fg; gv[gb + 256] = ct;
      gv[gb + 384] = cn; gv[gb + 512] = og; gv[gb + 640] = ht;
      pig[r] = (_Float16)ig; pfg[r] = (_Float16)fg; pct[r] = (_Float16)ct;
      pcn[r] = (_Float16)cn; pog[r] = (_Float16)og; pht[r] = (_Float16)ht;
      float gr = guar[r];
      atomicAdd(&rsh[hcol      ], ig * gr);
      atomicAdd(&rsh[hcol + 128], fg * gr);
      atomicAdd(&rsh[hcol + 256], ct * gr);
      atomicAdd(&rsh[hcol + 384], cn * gr);
      atomicAdd(&rsh[hcol + 512], og * gr);
      atomicAdd(&rsh[hcol + 640], ht * gr);
    }
    // transposed f16 copy for the Gram pass: gvT[gcol * B_ROWS + row], 16B/lane
    *(v8h*)&gvT[(size_t)(hcol      ) * B_ROWS + row0] = pig;
    *(v8h*)&gvT[(size_t)(hcol + 128) * B_ROWS + row0] = pfg;
    *(v8h*)&gvT[(size_t)(hcol + 256) * B_ROWS + row0] = pct;
    *(v8h*)&gvT[(size_t)(hcol + 384) * B_ROWS + row0] = pcn;
    *(v8h*)&gvT[(size_t)(hcol + 512) * B_ROWS + row0] = pog;
    *(v8h*)&gvT[(size_t)(hcol + 640) * B_ROWS + row0] = pht;
  }
  __syncthreads();
  for (int j = tid; j < GV_COLS; j += 256)
    rpart[(size_t)blockIdx.x * GV_COLS + j] = rsh[j];
}

// ---------------------------------------------------------------------------
// k2: Gram G = gv^T gv from the transposed f16 copy. Grid (21 blocks, 16
// row-chunks). Ping-pong LDS buffers filled by async global->LDS copies
// (ASYNCcnt) while WMMA consumes the previous chunk.
// ---------------------------------------------------------------------------
__global__ __launch_bounds__(256)
void k2_gram(const _Float16* __restrict__ gvT, float* __restrict__ gpart) {
  __shared__ _Float16 Ash[2][128 * K2_LDK];  // 2 x 10 KB
  __shared__ _Float16 Bsh[2][128 * K2_LDK];
  const int tid = threadIdx.x;
  const int lane = tid & 31, w = tid >> 5;
  const int kh = lane >> 4, nn = lane & 15;
  const int blk = blockIdx.x, chunk = blockIdx.y;
  const int bi = c_bi[blk], bj = c_bj[blk];
  const int r0 = chunk * (B_ROWS / S_CHUNKS);
  const int iters = (B_ROWS / S_CHUNKS) / 32;  // 128
  const int col = tid & 127;
  const int s0 = tid >> 7;  // 0 or 1

  v8f acc[8] = {};

  // stage one 32-row chunk (both 128-col groups) into buffer `buf`
  auto stage = [&](int buf, int rb) {
#pragma unroll
    for (int i = 0; i < 2; ++i) {
      int s = s0 + 2 * i;  // 4 x 16B segments per column
      const _Float16* ga = &gvT[(size_t)(bi * 128 + col) * B_ROWS + rb + s * 8];
      const _Float16* gb = &gvT[(size_t)(bj * 128 + col) * B_ROWS + rb + s * 8];
      async_copy16(ga, &Ash[buf][col * K2_LDK + s * 8]);
      async_copy16(gb, &Bsh[buf][col * K2_LDK + s * 8]);
    }
  };

  stage(0, r0);
  wait_async_all();
  __syncthreads();
  for (int it = 0; it < iters; ++it) {
    int buf = it & 1;
    if (it + 1 < iters) stage(buf ^ 1, r0 + (it + 1) * 32);
    v16h a = frag_k32(&Ash[buf][(w * 16 + nn) * K2_LDK], kh);
#pragma unroll
    for (int q = 0; q < 8; ++q) {
      v16h b = frag_k32(&Bsh[buf][(q * 16 + nn) * K2_LDK], kh);
      acc[q] = __builtin_amdgcn_wmma_f32_16x16x32_f16(
          false, a, false, b, (short)0, acc[q], false, false);
    }
    wait_async_all();
    __syncthreads();
  }

  size_t base = ((size_t)chunk * NBLK + blk) * 16384;
#pragma unroll
  for (int q = 0; q < 8; ++q) {
#pragma unroll
    for (int r = 0; r < 8; ++r) {
      int p = w * 16 + r + 8 * kh;
      int qq = q * 16 + nn;
      gpart[base + (size_t)p * 128 + qq] = acc[q][r];
    }
  }
}

// ---------------------------------------------------------------------------
// k3: reduce Gram partials over chunks, scatter into full symmetric G.
// ---------------------------------------------------------------------------
__global__ __launch_bounds__(256)
void k3_reduceG(const float* __restrict__ gpart, float* __restrict__ G) {
  int id = blockIdx.x * 256 + threadIdx.x;  // 21*16384
  if (id >= NBLK * 16384) return;
  int blk = id / 16384, e = id & 16383;
  float s = 0.0f;
#pragma unroll
  for (int ch = 0; ch < S_CHUNKS; ++ch)
    s += gpart[((size_t)ch * NBLK + blk) * 16384 + e];
  int p = e >> 7, q = e & 127;
  int P = c_bi[blk] * 128 + p, Q = c_bj[blk] * 128 + q;
  G[(size_t)P * GV_COLS + Q] = s;
  G[(size_t)Q * GV_COLS + P] = s;
}

__global__ __launch_bounds__(256)
void k3b_reduceR(const float* __restrict__ rpart, float* __restrict__ r) {
  int j = blockIdx.x * 256 + threadIdx.x;
  if (j >= GV_COLS) return;
  float s = 0.0f;
  for (int wg = 0; wg < 512; ++wg) s += rpart[(size_t)wg * GV_COLS + j];
  r[j] = s;
}

// ---------------------------------------------------------------------------
// k4: 13 GD steps on theta via the 768x768 Gram, then group-|theta| argmax.
// ---------------------------------------------------------------------------
__global__ __launch_bounds__(768)
void k4_gd(const float* __restrict__ G, const float* __restrict__ r,
           int* __restrict__ maxidx) {
  __shared__ float th[GV_COLS];
  __shared__ float red[GV_COLS];
  int j = threadIdx.x;
  th[j] = 1.0f;
  __syncthreads();
  const float coeff = 0.0001f * (2.0f / 65536.0f);
  for (int it = 0; it < GD_ITERS; ++it) {
    float dot = 0.0f;
    const float* gr = &G[(size_t)j * GV_COLS];
    for (int k = 0; k < GV_COLS; ++k) dot += gr[k] * th[k];
    float nt = th[j] - coeff * (dot - r[j]);
    __syncthreads();
    th[j] = nt;
    __syncthreads();
  }
  red[j] = fabsf(th[j]);
  __syncthreads();
  int gl = j & 127;
#pragma unroll
  for (int off = 64; off > 0; off >>= 1) {
    if (gl < off) red[j] += red[j + off];
    __syncthreads();
  }
  if (j == 0) {
    int best = 0; float bv = red[0];
#pragma unroll
    for (int g = 1; g < 6; ++g) {
      float v = red[g * 128];
      if (v > bv) { bv = v; best = g; }
    }
    maxidx[0] = best;
  }
}

// ---------------------------------------------------------------------------
// k5: new_h = selected candidate column-group of gv (f32 precision).
// ---------------------------------------------------------------------------
__global__ __launch_bounds__(256)
void k5_select(const float* __restrict__ gv, const int* __restrict__ maxidx,
               float* __restrict__ out) {
  size_t id = (size_t)blockIdx.x * 256 + threadIdx.x;
  int mi = maxidx[0];
  size_t row = id >> 7;
  int col = (int)(id & 127);
  __builtin_nontemporal_store(gv[row * GV_COLS + (size_t)mi * 128 + col], &out[id]);
}

// ---------------------------------------------------------------------------
extern "C" void kernel_launch(void* const* d_in, const int* in_sizes, int n_in,
                              void* d_out, int out_size, void* d_ws, size_t ws_size,
                              hipStream_t stream) {
  (void)in_sizes; (void)n_in; (void)out_size; (void)ws_size;
  const float* x    = (const float*)d_in[0];
  const float* h    = (const float*)d_in[1];
  const float* c    = (const float*)d_in[2];
  const float* W_ih = (const float*)d_in[3];
  const float* b_ih = (const float*)d_in[4];
  const float* W_hh = (const float*)d_in[5];
  float* out = (float*)d_out;

  char* ws = (char*)d_ws;
  size_t off = 0;
  auto alloc = [&](size_t bytes) -> void* {
    void* p = ws + off;
    off = (off + bytes + 255) & ~(size_t)255;
    return p;
  };
  _Float16* wpack = (_Float16*)alloc((size_t)131072 * 2);
  float*    gv    = (float*)alloc((size_t)B_ROWS * GV_COLS * 4);           // 192 MB
  _Float16* gvT   = (_Float16*)alloc((size_t)GV_COLS * B_ROWS * 2);        // 96 MB
  float*    gpart = (float*)alloc((size_t)S_CHUNKS * NBLK * 16384 * 4);    // 21 MB
  float*    rpart = (float*)alloc((size_t)512 * GV_COLS * 4);              // 1.5 MB
  float*    G     = (float*)alloc((size_t)GV_COLS * GV_COLS * 4);          // 2.25 MB
  float*    r     = (float*)alloc((size_t)GV_COLS * 4);
  int*      maxidx = (int*)alloc(256);

  k0_pack<<<512, 256, 0, stream>>>(W_ih, W_hh, wpack);
  k1_cell<<<B_ROWS / 128, 256, 0, stream>>>(x, h, c, b_ih, wpack,
                                            out + (size_t)B_ROWS * HDIM, gv, gvT, rpart);
  dim3 g2(NBLK, S_CHUNKS);
  k2_gram<<<g2, 256, 0, stream>>>(gvT, gpart);
  k3_reduceG<<<(NBLK * 16384 + 255) / 256, 256, 0, stream>>>(gpart, G);
  k3b_reduceR<<<3, 256, 0, stream>>>(rpart, r);
  k4_gd<<<1, 768, 0, stream>>>(G, r, maxidx);
  k5_select<<<(B_ROWS * HDIM) / 256, 256, 0, stream>>>(gv, maxidx, out);
}
